// CrossAttention_25589415150106
// MI455X (gfx1250) — matrix operand order repaired and checked by reading, not compiled, not run on hardware
//
#include <hip/hip_runtime.h>

typedef __attribute__((ext_vector_type(16))) _Float16 v16h;
typedef __attribute__((ext_vector_type(8)))  float    v8f;
typedef __attribute__((ext_vector_type(4)))  int      v4i;

#define B_   4
#define NQ_  2048
#define NC_  2048
#define H_   16
#define D_   64
#define SCALE_ 0.125f   // 64^-0.5 (folded into Q projection)

// ---- CDNA5 async global->LDS copy (ASYNCcnt) with safe fallback ------------
#if defined(__has_builtin)
#  if __has_builtin(__builtin_amdgcn_global_load_async_to_lds_b128)
#    define HAVE_ASYNC_LDS 1
#  endif
#  if __has_builtin(__builtin_amdgcn_s_wait_asynccnt)
#    define WAIT_ASYNC0() __builtin_amdgcn_s_wait_asynccnt(0)
#  endif
#endif
#ifndef HAVE_ASYNC_LDS
#  define HAVE_ASYNC_LDS 0
#endif
#ifndef WAIT_ASYNC0
#  define WAIT_ASYNC0() asm volatile("s_wait_asynccnt 0" ::: "memory")
#endif

typedef __attribute__((address_space(1))) v4i as1_v4i;
typedef __attribute__((address_space(3))) v4i as3_v4i;

__device__ __forceinline__ void lds_async_b128(void* lds_dst, const void* gsrc) {
#if HAVE_ASYNC_LDS
    __builtin_amdgcn_global_load_async_to_lds_b128(
        (as1_v4i*)(uintptr_t)gsrc,
        (as3_v4i*)(unsigned int)(uintptr_t)lds_dst,
        0, 0);
#else
    *(uint4*)lds_dst = *(const uint4*)gsrc;
#endif
}

// ---------------------------------------------------------------------------
// Kernel 1: C = A[Mx1024] * W[1024x1024] + bias, scaled, f16 output to
//   transposed==0 : dst[((b*16+h)*2048 + tok)*64 + d]   (Q / K layout)
//   transposed==1 : dst[((b*16+h)*64 + d)*2048 + tok]   (V^T layout)
// 256 threads (8 waves). Block tile 128x128, wave tile 64x32, BK=32.
// ---------------------------------------------------------------------------
__global__ __launch_bounds__(256) void qkv_proj_kernel(
    const float* __restrict__ A,
    const float* __restrict__ W,
    const float* __restrict__ bias,
    _Float16* __restrict__ dst,
    float outscale,
    int transposed)
{
    __shared__ _Float16 As[128][40];   // [m][k], 80B row stride (16B aligned)
    __shared__ _Float16 Bs[128][40];   // [n][k] (W tile transposed)

    const int tid  = threadIdx.x;
    const int lane = tid & 31;
    const int wave = tid >> 5;
    const int wm   = (wave >> 2) * 64;
    const int wn   = (wave & 3) * 32;
    const int bm   = blockIdx.y * 128;
    const int bn   = blockIdx.x * 128;
    const int lm   = lane & 15;
    const bool hi  = lane >= 16;

    v8f acc[4][2];
    #pragma unroll
    for (int i = 0; i < 4; i++)
        #pragma unroll
        for (int j = 0; j < 2; j++)
            acc[i][j] = (v8f){0.f,0.f,0.f,0.f,0.f,0.f,0.f,0.f};

    for (int k0 = 0; k0 < 1024; k0 += 32) {
        // prefetch next k-slice (global_prefetch_b8)
        if (k0 + 32 < 1024) {
            if (tid < 128)
                __builtin_prefetch(A + (size_t)(bm + tid) * 1024 + k0 + 32, 0, 1);
            else {
                int i = tid - 128;
                __builtin_prefetch(W + (size_t)(k0 + 32 + (i >> 2)) * 1024 + bn + (i & 3) * 32, 0, 1);
            }
        }
        // --- stage A tile 128x32 (f32 -> f16) ---
        #pragma unroll
        for (int i = 0; i < 4; i++) {
            int c   = tid + 256 * i;
            int row = c >> 3;
            int col = (c & 7) * 4;
            float4 f = *(const float4*)(A + (size_t)(bm + row) * 1024 + k0 + col);
            union { _Float16 h[4]; uint2 u; } pk;
            pk.h[0] = (_Float16)f.x; pk.h[1] = (_Float16)f.y;
            pk.h[2] = (_Float16)f.z; pk.h[3] = (_Float16)f.w;
            *(uint2*)&As[row][col] = pk.u;
        }
        // --- stage W tile 32x128 transposed -> Bs[n][k] ---
        #pragma unroll
        for (int i = 0; i < 4; i++) {
            int c  = tid + 256 * i;
            int kr = c >> 5;
            int n4 = (c & 31) * 4;
            float4 f = *(const float4*)(W + (size_t)(k0 + kr) * 1024 + bn + n4);
            Bs[n4 + 0][kr] = (_Float16)f.x;
            Bs[n4 + 1][kr] = (_Float16)f.y;
            Bs[n4 + 2][kr] = (_Float16)f.z;
            Bs[n4 + 3][kr] = (_Float16)f.w;
        }
        __syncthreads();

        v16h a[4], b[2];
        #pragma unroll
        for (int mi = 0; mi < 4; mi++) {
            const _Float16* p = &As[wm + mi * 16 + lm][hi ? 8 : 0];
            *(uint4*)&a[mi]         = *(const uint4*)p;
            *(((uint4*)&a[mi]) + 1) = *(const uint4*)(p + 16);
        }
        #pragma unroll
        for (int ni = 0; ni < 2; ni++) {
            const _Float16* p = &Bs[wn + ni * 16 + lm][hi ? 8 : 0];
            *(uint4*)&b[ni]         = *(const uint4*)p;
            *(((uint4*)&b[ni]) + 1) = *(const uint4*)(p + 16);
        }
        #pragma unroll
        for (int mi = 0; mi < 4; mi++)
            #pragma unroll
            for (int ni = 0; ni < 2; ni++)
                acc[mi][ni] = __builtin_amdgcn_wmma_f32_16x16x32_f16(
                    false, a[mi], false, b[ni], (short)0, acc[mi][ni], false, false);
        __syncthreads();
    }

    #pragma unroll
    for (int ni = 0; ni < 2; ni++) {
        int gn = bn + wn + ni * 16 + lm;
        float bv = bias[gn];
        int h = gn >> 6, d = gn & 63;
        #pragma unroll
        for (int mi = 0; mi < 4; mi++) {
            #pragma unroll
            for (int r = 0; r < 8; r++) {
                int gm  = bm + wm + mi * 16 + r + (hi ? 8 : 0);
                int bb  = gm >> 11, tok = gm & 2047;
                float v = (acc[mi][ni][r] + bv) * outscale;
                size_t idx = transposed
                    ? (((size_t)(bb * H_ + h)) * D_ + d) * (size_t)NC_ + tok
                    : (((size_t)(bb * H_ + h)) * (size_t)NC_ + tok) * D_ + d;
                dst[idx] = (_Float16)v;
            }
        }
    }
}

// ---------------------------------------------------------------------------
// Kernel 2: flash attention, double-buffered async K/V tiles.
// Block = (b,h, 128-query strip), 8 waves, each wave owns 16 query rows.
// ---------------------------------------------------------------------------
__global__ __launch_bounds__(256) void attn_kernel(
    const _Float16* __restrict__ Qf,          // [B,H,NQ,64]   (pre-scaled)
    const _Float16* __restrict__ Kf,          // [B,H,NC,64]
    const _Float16* __restrict__ Vt,          // [B,H,64,NC]
    const unsigned char* __restrict__ mask,   // [B,NC] bool
    float* __restrict__ out)                  // [B,NQ,1024]
{
    __shared__ _Float16 Ks[2][128][72];       // [key][d]  B-layout for QK^T
    __shared__ _Float16 VTs[2][64][136];      // [d][key]  B-layout for P*V
    __shared__ _Float16 Pb[8][16][40];        // per-wave P staging (32 keys)

    const int tid  = threadIdx.x;
    const int lane = tid & 31;
    const int wave = tid >> 5;
    const int lm   = lane & 15;
    const bool hi  = lane >= 16;
    const int bh   = blockIdx.y;
    const int b    = bh >> 4;
    const int h    = bh & 15;
    const int q0   = blockIdx.x * 128 + wave * 16;

    // Q fragments (16 rows x 64 dims), resident for the whole kernel
    v16h qf0, qf1;
    {
        const _Float16* p = Qf + (((size_t)bh) * NQ_ + q0) * D_ + lm * 64 + (hi ? 8 : 0);
        *(uint4*)&qf0         = *(const uint4*)p;
        *(((uint4*)&qf0) + 1) = *(const uint4*)(p + 16);
        *(uint4*)&qf1         = *(const uint4*)(p + 32);
        *(((uint4*)&qf1) + 1) = *(const uint4*)(p + 48);
    }

    v8f o[4];
    #pragma unroll
    for (int t = 0; t < 4; t++) o[t] = (v8f){0.f,0.f,0.f,0.f,0.f,0.f,0.f,0.f};
    float rowmax[8], rowsum[8];
    #pragma unroll
    for (int r = 0; r < 8; r++) { rowmax[r] = -3.4e38f; rowsum[r] = 0.f; }

    const unsigned char* mrow = mask + (size_t)b * NC_;
    const _Float16* Kbase = Kf + ((size_t)bh) * NC_ * D_;
    const _Float16* Vbase = Vt + ((size_t)bh) * D_ * NC_;

    // issue first tile (async global -> LDS, ASYNCcnt)
    {
        #pragma unroll
        for (int i = 0; i < 4; i++) {
            int c = tid + 256 * i;
            int row = c >> 3, q8 = (c & 7) * 8;
            lds_async_b128(&Ks[0][row][q8], Kbase + (size_t)row * 64 + q8);
        }
        #pragma unroll
        for (int i = 0; i < 4; i++) {
            int c = tid + 256 * i;
            int row = c >> 4, q8 = (c & 15) * 8;
            lds_async_b128(&VTs[0][row][q8], Vbase + (size_t)row * NC_ + q8);
        }
    }

    int cur = 0;
    for (int kt = 0; kt < NC_; kt += 128) {
        WAIT_ASYNC0();
        __syncthreads();   // tile resident; previous tile's readers all done

        if (kt + 128 < NC_) {   // prefetch next tile into the other buffer
            const _Float16* Kg = Kbase + (size_t)(kt + 128) * 64;
            const _Float16* Vg = Vbase + (kt + 128);
            #pragma unroll
            for (int i = 0; i < 4; i++) {
                int c = tid + 256 * i;
                int row = c >> 3, q8 = (c & 7) * 8;
                lds_async_b128(&Ks[cur ^ 1][row][q8], Kg + (size_t)row * 64 + q8);
            }
            #pragma unroll
            for (int i = 0; i < 4; i++) {
                int c = tid + 256 * i;
                int row = c >> 4, q8 = (c & 15) * 8;
                lds_async_b128(&VTs[cur ^ 1][row][q8], Vg + (size_t)row * NC_ + q8);
            }
        }

        _Float16 (*ks)[72]   = Ks[cur];
        _Float16 (*vts)[136] = VTs[cur];

        // ---- all 8 S subtiles of this 128-key tile (16 back-to-back WMMAs)
        v8f s[8];
        #pragma unroll
        for (int j = 0; j < 8; j++) {
            v16h kf0, kf1;
            const _Float16* p = &ks[j * 16 + lm][hi ? 8 : 0];
            *(uint4*)&kf0         = *(const uint4*)p;
            *(((uint4*)&kf0) + 1) = *(const uint4*)(p + 16);
            *(uint4*)&kf1         = *(const uint4*)(p + 32);
            *(((uint4*)&kf1) + 1) = *(const uint4*)(p + 48);
            s[j] = (v8f){0.f,0.f,0.f,0.f,0.f,0.f,0.f,0.f};
            s[j] = __builtin_amdgcn_wmma_f32_16x16x32_f16(false, qf0, false, kf0, (short)0, s[j], false, false);
            s[j] = __builtin_amdgcn_wmma_f32_16x16x32_f16(false, qf1, false, kf1, (short)0, s[j], false, false);
        }

        float madd[8];
        #pragma unroll
        for (int j = 0; j < 8; j++)
            madd[j] = mrow[kt + j * 16 + lm] ? 0.0f : -3.0e38f;

        // ---- one softmax pass per row over the whole 128-key tile
        #pragma unroll
        for (int r = 0; r < 8; r++) {
            float m = -3.4e38f;
            #pragma unroll
            for (int j = 0; j < 8; j++) m = fmaxf(m, s[j][r] + madd[j]);
            m = fmaxf(m, __shfl_xor(m, 1, 32));
            m = fmaxf(m, __shfl_xor(m, 2, 32));
            m = fmaxf(m, __shfl_xor(m, 4, 32));
            m = fmaxf(m, __shfl_xor(m, 8, 32));
            float nm   = fmaxf(rowmax[r], m);
            float corr = (nm > -1.0e37f) ? __expf(rowmax[r] - nm) : 1.0f;
            float lsum = 0.f;
            #pragma unroll
            for (int j = 0; j < 8; j++) {
                float e = __expf(s[j][r] + madd[j] - nm);
                lsum += e;
                s[j][r] = e;                      // keep P in the S registers
            }
            lsum += __shfl_xor(lsum, 1, 32);
            lsum += __shfl_xor(lsum, 2, 32);
            lsum += __shfl_xor(lsum, 4, 32);
            lsum += __shfl_xor(lsum, 8, 32);
            rowsum[r] = rowsum[r] * corr + lsum;
            rowmax[r] = nm;
            #pragma unroll
            for (int t = 0; t < 4; t++) o[t][r] *= corr;
        }

        // ---- P @ V per 32-key chunk (stage P through per-wave LDS)
        #pragma unroll
        for (int kk = 0; kk < 4; kk++) {
            #pragma unroll
            for (int jj = 0; jj < 2; jj++) {
                int j = kk * 2 + jj;
                #pragma unroll
                for (int r = 0; r < 8; r++)
                    Pb[wave][r + (hi ? 8 : 0)][jj * 16 + lm] = (_Float16)s[j][r];
            }
            asm volatile("s_wait_dscnt 0" ::: "memory");  // per-wave LDS RAW
            v16h pf;
            const _Float16* pp = &Pb[wave][lm][hi ? 8 : 0];
            *(uint4*)&pf         = *(const uint4*)pp;
            *(((uint4*)&pf) + 1) = *(const uint4*)(pp + 16);
            #pragma unroll
            for (int t = 0; t < 4; t++) {
                v16h vf;
                const _Float16* pv2 = &vts[t * 16 + lm][kk * 32 + (hi ? 8 : 0)];
                *(uint4*)&vf         = *(const uint4*)pv2;
                *(((uint4*)&vf) + 1) = *(const uint4*)(pv2 + 16);
                o[t] = __builtin_amdgcn_wmma_f32_16x16x32_f16(false, pf, false, vf, (short)0, o[t], false, false);
            }
        }
        cur ^= 1;
    }

    #pragma unroll
    for (int r = 0; r < 8; r++) {
        float inv = 1.0f / rowsum[r];
        int q = q0 + r + (hi ? 8 : 0);
        float* op = out + ((size_t)b * NQ_ + q) * 1024 + h * 64;
        #pragma unroll
        for (int t = 0; t < 4; t++)
            op[t * 16 + lm] = o[t][r] * inv;
    }
}

// ---------------------------------------------------------------------------
extern "C" void kernel_launch(void* const* d_in, const int* in_sizes, int n_in,
                              void* d_out, int out_size, void* d_ws, size_t ws_size,
                              hipStream_t stream) {
    (void)in_sizes; (void)n_in; (void)out_size; (void)ws_size;
    const float* x   = (const float*)d_in[0];
    const float* ctx = (const float*)d_in[1];
    const unsigned char* mask = (const unsigned char*)d_in[2];
    const float* Wq = (const float*)d_in[3];
    const float* bq = (const float*)d_in[4];
    const float* Wk = (const float*)d_in[5];
    const float* bk = (const float*)d_in[6];
    const float* Wv = (const float*)d_in[7];
    const float* bv = (const float*)d_in[8];
    float* out = (float*)d_out;

    const size_t elems = (size_t)B_ * H_ * NQ_ * D_;   // 8M halves = 16 MB
    _Float16* Qf = (_Float16*)d_ws;
    _Float16* Kf = Qf + elems;
    _Float16* Vt = Kf + elems;

    dim3 blk(256);
    dim3 g1(1024 / 128, (B_ * NQ_) / 128);   // (8, 64)
    hipLaunchKernelGGL(qkv_proj_kernel, g1, blk, 0, stream, x,   Wq, bq, Qf, SCALE_, 0);
    hipLaunchKernelGGL(qkv_proj_kernel, g1, blk, 0, stream, ctx, Wk, bk, Kf, 1.0f,  0);
    hipLaunchKernelGGL(qkv_proj_kernel, g1, blk, 0, stream, ctx, Wv, bv, Vt, 1.0f,  1);

    dim3 g2(NQ_ / 128, B_ * H_);             // (16, 64)
    hipLaunchKernelGGL(attn_kernel, g2, blk, 0, stream, Qf, Kf, Vt, mask, out);
}
